// Seq2seq_12850542150593
// MI455X (gfx1250) — matrix operand order repaired and checked by reading, not compiled, not run on hardware
//
#include <hip/hip_runtime.h>
#include <hip/hip_bf16.h>

typedef __bf16 bf16_t;
typedef __attribute__((ext_vector_type(16))) __bf16 v16bf;
typedef __attribute__((ext_vector_type(8)))  float  v8f;

#define BB 16
#define HH 512
#define AA 32
#define LL 64
#define TT 63
#define VV 32000
#define NPART 250        // 250 * 128 = 32000
#define FRAG 512         // elements per 16x16x32 bf16 fragment (32 lanes * 16)

// ---------- fragment helpers ----------
// A fragment layout (ISA 7.12.2 16-bit A): lane = m + 16*half, where for
// K-local r (0..31): half = (r>>3)&1, e = (r&7) | ((r&16)>>1).
// B fragment layout: same K mapping as A: lane = n + 16*half, e <-> K = (e&8)*2 + (lane>>4)*8 + (e&7).
// Stored as contiguous [lane][e] blocks of 1KB -> one 32B load per lane.

static __device__ inline v16bf load_frag(const bf16_t* p) {
    return ((const v16bf*)p)[threadIdx.x & 31];
}

static __device__ inline void store_a_frag(bf16_t* dst, int kt, int m, int r, bf16_t v) {
    int lane = m + (((r >> 3) & 1) << 4);
    int e = (r & 7) | ((r & 16) >> 1);
    dst[((size_t)kt * 32 + lane) * 16 + e] = v;
}

static __device__ inline v8f wmma_bf16(v16bf a, v16bf b, v8f c) {
    return __builtin_amdgcn_wmma_f32_16x16x32_bf16(
        false, a, false, b, (short)0, c, false, false);
}

static __device__ inline float sigmoidf_(float x) { return 1.0f / (1.0f + __expf(-x)); }

// ---------- init ----------
__global__ void k_init(float* hs, float* h, float* c, bf16_t* h_fr, float* loss) {
    int i = blockIdx.x * blockDim.x + threadIdx.x;
    int n = BB * (AA + LL) * HH;
    for (int j = i; j < n; j += gridDim.x * blockDim.x) hs[j] = 0.0f;
    if (i < BB * HH) { h[i] = 0.0f; c[i] = 0.0f; h_fr[i] = (bf16_t)0.0f; }
    if (i == 0) *loss = 0.0f;
}

// ---------- fp32 row-major (K x N) -> bf16 B-fragment order ----------
// dst index: (((nt*Ktiles)+kt)*32 + lane)*16 + e ; src K = kt*32+(e&8)*2+(lane>>4)*8+(e&7), n = nt*16+(lane&15)
__global__ void k_pack_b(const float* __restrict__ src, bf16_t* __restrict__ dst, int K, int N) {
    int i = blockIdx.x * blockDim.x + threadIdx.x;
    if (i >= K * N) return;
    int Ktiles = K >> 5;
    int nt  = i / (Ktiles * FRAG);
    int rem = i - nt * (Ktiles * FRAG);
    int kt  = rem >> 9;
    int j   = rem & (FRAG - 1);
    int lane = j >> 4, e = j & 15;
    int n  = nt * 16 + (lane & 15);
    int kg = kt * 32 + ((e & 8) << 1) + ((lane >> 4) << 3) + (e & 7);
    dst[i] = (bf16_t)src[(size_t)kg * N + n];
}

// ---------- recurrent cell: attention + SELU MLP + LSTM (1 block, 16 waves) ----------
__global__ __launch_bounds__(512)
void k_rstep(const int* __restrict__ s_tensor, const int* __restrict__ parents,
             const float* __restrict__ emb,
             const float* __restrict__ b_hid, const float* __restrict__ v_w,
             const float* __restrict__ v_b,   const float* __restrict__ b_ctx,
             const float* __restrict__ b_lstm,const float* __restrict__ w_sw,
             const float* __restrict__ b_sw,
             const bf16_t* __restrict__ Whid_fr, const bf16_t* __restrict__ Wctx_fr,
             const bf16_t* __restrict__ Wih_fr,  const bf16_t* __restrict__ Whh_fr,
             float* hs_buf, float* h, float* c, bf16_t* h_fr,
             float* q, bf16_t* xc_fr, bf16_t* hid_fr, float* gates,
             bf16_t* zcat_fr, float* s_sw, int t)
{
    const int wid  = threadIdx.x >> 5;
    const int lane = threadIdx.x & 31;
    const int n16  = lane & 15;
    const int half = lane >> 4;

    // ---- Phase 1: q = h @ W_hid + b_hid  (32 N-tiles, 2 per wave; Ktiles=16)
#pragma unroll
    for (int tt = 0; tt < 2; ++tt) {
        int nt = wid * 2 + tt;
        v8f acc = {};
#pragma unroll 4
        for (int kt = 0; kt < 16; ++kt) {
            v16bf a = load_frag(h_fr + kt * FRAG);
            v16bf b = load_frag(Whid_fr + ((size_t)nt * 16 + kt) * FRAG);
            acc = wmma_bf16(a, b, acc);
        }
#pragma unroll
        for (int v = 0; v < 8; ++v) {
            int m = v + 8 * half, col = nt * 16 + n16;
            q[m * HH + col] = acc[v] + b_hid[col];
        }
    }
    __threadfence(); __syncthreads();

    // ---- Phase 2: attention (wave w -> batch row w, lane -> attention slot)
    {
        int b_row = wid, a_idx = lane;
        const float4* m4 = (const float4*)(hs_buf + ((size_t)b_row * (AA + LL) + (t + a_idx)) * HH);
        const float4* q4 = (const float4*)(q + b_row * HH);
        const float4* v4 = (const float4*)v_w;
        float sc = 0.0f;
        for (int k = 0; k < HH / 4; ++k) {
            float4 qq = q4[k], mm = m4[k], vv = v4[k];
            sc += tanhf(qq.x + mm.x) * vv.x + tanhf(qq.y + mm.y) * vv.y
                + tanhf(qq.z + mm.z) * vv.z + tanhf(qq.w + mm.w) * vv.w;
        }
        sc += v_b[0];
        if (a_idx < AA - t) sc = -1e20f;
        float mx = sc;
        for (int o = 16; o > 0; o >>= 1) mx = fmaxf(mx, __shfl_xor(mx, o, 32));
        float e = __expf(sc - mx);
        float ssum = e;
        for (int o = 16; o > 0; o >>= 1) ssum += __shfl_xor(ssum, o, 32);
        float attn = e / ssum;

        int tok = (t == 0) ? 2 : s_tensor[b_row * LL + (t - 1)];   // SOS=2
        const float* xrow = emb + (size_t)tok * HH;
#pragma unroll 1
        for (int j = 0; j < HH / 32; ++j) {
            int k = lane + 32 * j;
            float ctxv = 0.0f;
            for (int a2 = 0; a2 < AA; ++a2) {
                float w = __shfl(attn, a2, 32);
                ctxv += w * hs_buf[((size_t)b_row * (AA + LL) + (t + a2)) * HH + k];
            }
            store_a_frag(xc_fr, k >> 5, b_row, k & 31, (bf16_t)xrow[k]);              // x part
            store_a_frag(xc_fr, (HH + k) >> 5, b_row, k & 31, (bf16_t)ctxv);          // ctx part
        }
    }
    __threadfence(); __syncthreads();

    // ---- Phase 3: hid = selu([x,ctx] @ W_ctx + b_ctx)   Ktiles=32, 32 N-tiles
#pragma unroll
    for (int tt = 0; tt < 2; ++tt) {
        int nt = wid * 2 + tt;
        v8f acc = {};
#pragma unroll 4
        for (int kt = 0; kt < 32; ++kt) {
            v16bf a = load_frag(xc_fr + kt * FRAG);
            v16bf b = load_frag(Wctx_fr + ((size_t)nt * 32 + kt) * FRAG);
            acc = wmma_bf16(a, b, acc);
        }
#pragma unroll
        for (int v = 0; v < 8; ++v) {
            int m = v + 8 * half, col = nt * 16 + n16;
            float x = acc[v] + b_ctx[col];
            float r = (x > 0.0f) ? x : 1.6732632423543772f * (__expf(x) - 1.0f);
            store_a_frag(hid_fr, col >> 5, m, col & 31, (bf16_t)(1.0507009873554805f * r));
        }
    }
    __threadfence(); __syncthreads();

    // ---- Phase 4: gates = hid @ W_ih + h @ W_hh + b_lstm   128 N-tiles, 8 per wave
#pragma unroll 1
    for (int tt = 0; tt < 8; ++tt) {
        int nt = wid * 8 + tt;
        v8f acc = {};
#pragma unroll 4
        for (int kt = 0; kt < 16; ++kt) {
            v16bf a = load_frag(hid_fr + kt * FRAG);
            v16bf b = load_frag(Wih_fr + ((size_t)nt * 16 + kt) * FRAG);
            acc = wmma_bf16(a, b, acc);
            a = load_frag(h_fr + kt * FRAG);
            b = load_frag(Whh_fr + ((size_t)nt * 16 + kt) * FRAG);
            acc = wmma_bf16(a, b, acc);
        }
#pragma unroll
        for (int v = 0; v < 8; ++v) {
            int m = v + 8 * half, col = nt * 16 + n16;
            gates[m * 4 * HH + col] = acc[v] + b_lstm[col];
        }
    }
    __threadfence(); __syncthreads();

    // ---- Phase 5: LSTM elementwise, hs_buf / fragment updates
    {
        int p = parents[t];
        int hi = (t - 1 > 0) ? (t - 1) : 0;
        if (p < 0) p = 0; if (p > hi) p = hi;
        for (int idx = threadIdx.x; idx < BB * HH; idx += 512) {
            int b_row = idx >> 9, k = idx & (HH - 1);
            const float* g = gates + b_row * 4 * HH;
            float gi = g[k], gf = g[HH + k], gg = g[2 * HH + k], go = g[3 * HH + k];
            float cn = sigmoidf_(gf) * c[idx] + sigmoidf_(gi) * tanhf(gg);
            float hn = sigmoidf_(go) * tanhf(cn);
            float hp = hs_buf[((size_t)b_row * (AA + LL) + AA + p) * HH + k]; // read before write
            c[idx] = cn; h[idx] = hn;
            hs_buf[((size_t)b_row * (AA + LL) + AA + t) * HH + k] = hn;
            store_a_frag(h_fr, k >> 5, b_row, k & 31, (bf16_t)hn);
            store_a_frag(zcat_fr, k >> 5, b_row, k & 31, (bf16_t)hn);
            store_a_frag(zcat_fr, (HH + k) >> 5, b_row, k & 31, (bf16_t)cn);
            store_a_frag(zcat_fr, (2 * HH + k) >> 5, b_row, k & 31, (bf16_t)hp);
        }
    }
    __threadfence(); __syncthreads();

    // ---- Phase 6: switch gate s = sigmoid([h,c] . w_sw + b_sw), wave w -> row w
    {
        int b_row = wid;
        float acc = 0.0f;
        for (int k = lane; k < HH; k += 32)
            acc += h[b_row * HH + k] * w_sw[k] + c[b_row * HH + k] * w_sw[HH + k];
        for (int o = 16; o > 0; o >>= 1) acc += __shfl_xor(acc, o, 32);
        if (lane == 0) s_sw[b_row] = sigmoidf_(acc + b_sw[0]);
    }
}

// ---------- vocabulary GEMM + online softmax partials (250 blocks x 8 waves) ----------
__global__ __launch_bounds__(256)
void k_logits(const bf16_t* __restrict__ zcat_fr, const bf16_t* __restrict__ Wg_fr,
              const float* __restrict__ b_glob, const int* __restrict__ s_tensor,
              float* pmax, float* psum, float* pamaxv, int* pamaxi, float* ptgt, int t)
{
    __shared__ float lds_logits[BB * 128];
    const int wid  = threadIdx.x >> 5;
    const int lane = threadIdx.x & 31;
    const int n16  = lane & 15;
    const int half = lane >> 4;
    const int colB = blockIdx.x * 128;
    const int ntg  = blockIdx.x * 8 + wid;           // global N-tile (0..1999)

    const bf16_t* bbase = Wg_fr + (size_t)ntg * 48 * FRAG;   // 48KB contiguous per wave
    v8f acc = {};
#pragma unroll 4
    for (int kt = 0; kt < 48; ++kt) {
        __builtin_prefetch(bbase + (size_t)(kt + 4) * FRAG, 0, 1);
        v16bf a = load_frag(zcat_fr + kt * FRAG);
        v16bf b = load_frag(bbase + (size_t)kt * FRAG);
        acc = wmma_bf16(a, b, acc);
    }
#pragma unroll
    for (int v = 0; v < 8; ++v) {
        int m = v + 8 * half;
        lds_logits[m * 128 + wid * 16 + n16] = acc[v] + b_glob[ntg * 16 + n16];
    }
    __syncthreads();

#pragma unroll
    for (int rr = 0; rr < 2; ++rr) {
        int m = wid + rr * 8;
        float vals[4];
        float mx = -3.4e38f, bestv = -3.4e38f; int besti = 0x7fffffff;
#pragma unroll
        for (int j = 0; j < 4; ++j) {
            int cc = lane * 4 + j;
            float x = lds_logits[m * 128 + cc];
            vals[j] = x;
            mx = fmaxf(mx, x);
            if (x > bestv) { bestv = x; besti = colB + cc; }
        }
        for (int o = 16; o > 0; o >>= 1) {
            mx = fmaxf(mx, __shfl_xor(mx, o, 32));
            float ov = __shfl_xor(bestv, o, 32);
            int   oi = __shfl_xor(besti, o, 32);
            if (ov > bestv || (ov == bestv && oi < besti)) { bestv = ov; besti = oi; }
        }
        float se = 0.0f;
#pragma unroll
        for (int j = 0; j < 4; ++j) se += __expf(vals[j] - mx);
        for (int o = 16; o > 0; o >>= 1) se += __shfl_xor(se, o, 32);

        if (lane == 0) {
            int tgt = s_tensor[m * LL + (t + 1)];
            float tl = 0.0f;
            if (tgt >= colB && tgt < colB + 128) tl = lds_logits[m * 128 + (tgt - colB)];
            int pi = blockIdx.x * BB + m;
            pmax[pi] = mx; psum[pi] = se; pamaxv[pi] = bestv; pamaxi[pi] = besti; ptgt[pi] = tl;
        }
    }
}

// ---------- cross-block softmax/argmax reduce, nll + pred (1 block, 16 waves) ----------
__global__ __launch_bounds__(512)
void k_reduce(const float* __restrict__ pmax, const float* __restrict__ psum,
              const float* __restrict__ pamaxv, const int* __restrict__ pamaxi,
              const float* __restrict__ ptgt, const float* __restrict__ s_sw,
              const int* __restrict__ s_tensor, float* loss_acc, float* d_out, int t)
{
    int b = threadIdx.x >> 5, lane = threadIdx.x & 31;
    float m = -3.4e38f, s = 0.0f, bv = -3.4e38f; int bi = 0x7fffffff;
    for (int p = lane; p < NPART; p += 32) {
        int pi = p * BB + b;
        float pm = pmax[pi], ps = psum[pi];
        float nm = fmaxf(m, pm);
        s = s * __expf(m - nm) + ps * __expf(pm - nm);
        m = nm;
        float pv = pamaxv[pi]; int pii = pamaxi[pi];
        if (pv > bv || (pv == bv && pii < bi)) { bv = pv; bi = pii; }
    }
    for (int o = 16; o > 0; o >>= 1) {
        float om = __shfl_xor(m, o, 32), os = __shfl_xor(s, o, 32);
        float nm = fmaxf(m, om);
        s = s * __expf(m - nm) + os * __expf(om - nm);
        m = nm;
        float ov = __shfl_xor(bv, o, 32); int oi = __shfl_xor(bi, o, 32);
        if (ov > bv || (ov == bv && oi < bi)) { bv = ov; bi = oi; }
    }
    if (lane == 0) {
        int tgt = s_tensor[b * LL + (t + 1)];
        float logZ = m + __logf(s);
        float tl = ptgt[(tgt / 128) * BB + b];
        float nll = (tgt != 1) ? (-s_sw[b] * (tl - logZ)) : 0.0f;   // PAD=1
        atomicAdd(loss_acc, nll);
        d_out[1 + b * TT + t] = (float)bi;   // preds.T flattened (B, T)
    }
}

__global__ void k_final(const float* loss_acc, float* d_out) {
    d_out[0] = loss_acc[0] / (float)BB;
}

// ---------- host ----------
extern "C" void kernel_launch(void* const* d_in, const int* in_sizes, int n_in,
                              void* d_out, int out_size, void* d_ws, size_t ws_size,
                              hipStream_t stream) {
    const int*   s_tensor = (const int*)  d_in[0];
    const int*   parents  = (const int*)  d_in[1];
    const float* emb      = (const float*)d_in[2];
    const float* W_hid    = (const float*)d_in[3];
    const float* b_hid    = (const float*)d_in[4];
    const float* v_w      = (const float*)d_in[5];
    const float* v_b      = (const float*)d_in[6];
    const float* W_ctx    = (const float*)d_in[7];
    const float* b_ctx    = (const float*)d_in[8];
    const float* W_ih     = (const float*)d_in[9];
    const float* W_hh     = (const float*)d_in[10];
    const float* b_lstm   = (const float*)d_in[11];
    const float* W_glob   = (const float*)d_in[12];
    const float* b_glob   = (const float*)d_in[13];
    const float* w_sw     = (const float*)d_in[14];
    const float* b_sw     = (const float*)d_in[15];

    char* w = (char*)d_ws;
    auto alloc = [&](size_t bytes) -> void* {
        void* p = (void*)w; w += (bytes + 1023) & ~(size_t)1023; return p;
    };
    bf16_t* Wg_fr   = (bf16_t*)alloc((size_t)3 * HH * VV * sizeof(bf16_t));   // 98.3 MB, L2-resident
    bf16_t* Wih_fr  = (bf16_t*)alloc((size_t)HH * 4 * HH * sizeof(bf16_t));
    bf16_t* Whh_fr  = (bf16_t*)alloc((size_t)HH * 4 * HH * sizeof(bf16_t));
    bf16_t* Wctx_fr = (bf16_t*)alloc((size_t)2 * HH * HH * sizeof(bf16_t));
    bf16_t* Whid_fr = (bf16_t*)alloc((size_t)HH * HH * sizeof(bf16_t));
    float*  hs_buf  = (float*) alloc((size_t)BB * (AA + LL) * HH * sizeof(float));
    float*  h       = (float*) alloc((size_t)BB * HH * sizeof(float));
    float*  c       = (float*) alloc((size_t)BB * HH * sizeof(float));
    bf16_t* h_fr    = (bf16_t*)alloc((size_t)BB * HH * sizeof(bf16_t));
    float*  q       = (float*) alloc((size_t)BB * HH * sizeof(float));
    bf16_t* xc_fr   = (bf16_t*)alloc((size_t)BB * 2 * HH * sizeof(bf16_t));
    bf16_t* hid_fr  = (bf16_t*)alloc((size_t)BB * HH * sizeof(bf16_t));
    float*  gates   = (float*) alloc((size_t)BB * 4 * HH * sizeof(float));
    bf16_t* zcat_fr = (bf16_t*)alloc((size_t)BB * 3 * HH * sizeof(bf16_t));
    float*  s_sw    = (float*) alloc((size_t)BB * sizeof(float));
    float*  pmaxb   = (float*) alloc((size_t)NPART * BB * sizeof(float));
    float*  psumb   = (float*) alloc((size_t)NPART * BB * sizeof(float));
    float*  pamaxv  = (float*) alloc((size_t)NPART * BB * sizeof(float));
    int*    pamaxi  = (int*)   alloc((size_t)NPART * BB * sizeof(int));
    float*  ptgt    = (float*) alloc((size_t)NPART * BB * sizeof(float));
    float*  loss    = (float*) alloc(1024);

    float* out = (float*)d_out;

    k_init<<<3072, 256, 0, stream>>>(hs_buf, h, c, h_fr, loss);

    int nWg = 3 * HH * VV;
    k_pack_b<<<(nWg + 255) / 256, 256, 0, stream>>>(W_glob, Wg_fr, 3 * HH, VV);
    k_pack_b<<<(HH * 4 * HH + 255) / 256, 256, 0, stream>>>(W_ih, Wih_fr, HH, 4 * HH);
    k_pack_b<<<(HH * 4 * HH + 255) / 256, 256, 0, stream>>>(W_hh, Whh_fr, HH, 4 * HH);
    k_pack_b<<<(2 * HH * HH + 255) / 256, 256, 0, stream>>>(W_ctx, Wctx_fr, 2 * HH, HH);
    k_pack_b<<<(HH * HH + 255) / 256, 256, 0, stream>>>(W_hid, Whid_fr, HH, HH);

    for (int t = 0; t < TT; ++t) {
        k_rstep<<<1, 512, 0, stream>>>(s_tensor, parents, emb,
                                       b_hid, v_w, v_b, b_ctx, b_lstm, w_sw, b_sw,
                                       Whid_fr, Wctx_fr, Wih_fr, Whh_fr,
                                       hs_buf, h, c, h_fr, q, xc_fr, hid_fr, gates,
                                       zcat_fr, s_sw, t);
        k_logits<<<NPART, 256, 0, stream>>>(zcat_fr, Wg_fr, b_glob, s_tensor,
                                            pmaxb, psumb, pamaxv, pamaxi, ptgt, t);
        k_reduce<<<1, 512, 0, stream>>>(pmaxb, psumb, pamaxv, pamaxi, ptgt,
                                        s_sw, s_tensor, loss, out, t);
    }
    k_final<<<1, 1, 0, stream>>>(loss, out);
}
